// CoreferenceCrossEncoder_6760278524356
// MI455X (gfx1250) — matrix operand order repaired and checked by reading, not compiled, not run on hardware
//
#include <hip/hip_runtime.h>
#include <math.h>

// ---------------- model dimensions ----------------
constexpr int Bb  = 2;
constexpr int Ssz = 4096;
constexpr int Dm  = 768;
constexpr int Hh  = 12;
constexpr int DHh = 64;
constexpr int NLn = 2;
constexpr int WIN = 256;     // attention window
constexpr int Gg  = 4;       // global tokens
constexpr int FFd = 3072;
constexpr int NC  = Ssz / WIN;     // 16 chunks
constexpr int BS  = Bb * Ssz;      // 8192 token rows

// ---------------- WMMA types (gfx1250, wave32) ----------------
typedef __bf16 bf16;
typedef __attribute__((ext_vector_type(16))) __bf16 v16bf;
typedef __attribute__((ext_vector_type(8)))  float  v8f;

__device__ __forceinline__ float gelu_f(float x) {
  const float c = 0.7978845608028654f;
  return 0.5f * x * (1.0f + tanhf(c * (x + 0.044715f * x * x * x)));
}

// =====================================================================
// Embedding gather + LayerNorm.  grid = BS rows, 256 threads (3 elem/thr)
// =====================================================================
__global__ __launch_bounds__(256)
void k_embed_ln(const int* __restrict__ sent,
                const float* __restrict__ wemb, const float* __restrict__ pemb,
                const float* __restrict__ temb,
                const float* __restrict__ gam, const float* __restrict__ bet,
                float* __restrict__ x)
{
  __shared__ float red[256];
  const int row = blockIdx.x;
  const int s   = row % Ssz;
  const int tok = sent[row];
  const int tid = threadIdx.x;

  float e[3]; float sum = 0.f;
  #pragma unroll
  for (int i = 0; i < 3; ++i) {
    int d = tid + i * 256;
    e[i] = wemb[(size_t)tok * Dm + d] + pemb[(size_t)s * Dm + d] + temb[d];
    sum += e[i];
  }
  red[tid] = sum; __syncthreads();
  for (int o = 128; o > 0; o >>= 1) { if (tid < o) red[tid] += red[tid + o]; __syncthreads(); }
  const float mean = red[0] / (float)Dm;
  __syncthreads();
  float sq = 0.f;
  #pragma unroll
  for (int i = 0; i < 3; ++i) { float t = e[i] - mean; sq += t * t; }
  red[tid] = sq; __syncthreads();
  for (int o = 128; o > 0; o >>= 1) { if (tid < o) red[tid] += red[tid + o]; __syncthreads(); }
  const float rstd = rsqrtf(red[0] / (float)Dm + 1e-5f);
  #pragma unroll
  for (int i = 0; i < 3; ++i) {
    int d = tid + i * 256;
    x[(size_t)row * Dm + d] = (e[i] - mean) * rstd * gam[d] + bet[d];
  }
}

// =====================================================================
// x = LN(x + y).  grid = BS rows, 256 threads
// =====================================================================
__global__ __launch_bounds__(256)
void k_ln_resid(float* __restrict__ x, const float* __restrict__ y,
                const float* __restrict__ gam, const float* __restrict__ bet)
{
  __shared__ float red[256];
  const int row = blockIdx.x;
  const int tid = threadIdx.x;
  float e[3]; float sum = 0.f;
  #pragma unroll
  for (int i = 0; i < 3; ++i) {
    int d = tid + i * 256;
    e[i] = x[(size_t)row * Dm + d] + y[(size_t)row * Dm + d];
    sum += e[i];
  }
  red[tid] = sum; __syncthreads();
  for (int o = 128; o > 0; o >>= 1) { if (tid < o) red[tid] += red[tid + o]; __syncthreads(); }
  const float mean = red[0] / (float)Dm;
  __syncthreads();
  float sq = 0.f;
  #pragma unroll
  for (int i = 0; i < 3; ++i) { float t = e[i] - mean; sq += t * t; }
  red[tid] = sq; __syncthreads();
  for (int o = 128; o > 0; o >>= 1) { if (tid < o) red[tid] += red[tid + o]; __syncthreads(); }
  const float rstd = rsqrtf(red[0] / (float)Dm + 1e-5f);
  __syncthreads();
  #pragma unroll
  for (int i = 0; i < 3; ++i) {
    int d = tid + i * 256;
    x[(size_t)row * Dm + d] = (e[i] - mean) * rstd * gam[d] + bet[d];
  }
}

// =====================================================================
// WMMA GEMM:  C[M,N] = act(scale * (A[M,K] @ W[K,N] + bias))
// Block tile 128x128, 8 wave32 waves, each wave: 32x64 (2x4 WMMA tiles).
// Double-buffered LDS staging (one barrier per k-step) + global prefetch
// of the k+64 tiles.  fp32 -> bf16 conversion on the staging path;
// v_wmma_f32_16x16x32_bf16 with fp32 accumulation.
// grid = (N/128, M/128), 256 threads.
// =====================================================================
__global__ __launch_bounds__(256)
void k_gemm(const float* __restrict__ A, const float* __restrict__ Wd,
            const float* __restrict__ bias, float* __restrict__ C,
            int M, int N, int K, float scale, int act)
{
  __shared__ bf16 As[2][128][40];   // [buf][m][k], rows padded to spread banks
  __shared__ bf16 Bs[2][128][40];   // [buf][n][k] (transposed)

  const int tid  = threadIdx.x;
  const int lane = tid & 31;
  const int w    = tid >> 5;
  const int wm   = w & 3;     // wave row   -> m offset 32*wm
  const int wn   = w >> 2;    // wave col   -> n offset 64*wn
  const int m0   = blockIdx.y * 128;
  const int n0   = blockIdx.x * 128;
  const int hi   = lane >> 4;   // half-wave selects K sub-range
  const int lr   = lane & 15;

  const v8f vz = {0.f,0.f,0.f,0.f,0.f,0.f,0.f,0.f};
  v8f acc[2][4];
  #pragma unroll
  for (int i = 0; i < 2; ++i)
    #pragma unroll
    for (int j = 0; j < 4; ++j) acc[i][j] = vz;

  union Frag { v16bf v; unsigned u[8]; };

  // ---- tile staging (fp32 global -> bf16 LDS) ----
  auto stage = [&](int buf, int kbase) {
    #pragma unroll
    for (int it = 0; it < 4; ++it) {
      int cc  = tid + it * 256;            // 1024 float4 chunks for A
      int row = cc >> 3;
      int kk  = (cc & 7) << 2;
      const float4 f = *reinterpret_cast<const float4*>(
          A + (size_t)(m0 + row) * K + kbase + kk);
      bf16* dst = &As[buf][row][kk];
      dst[0] = (bf16)f.x; dst[1] = (bf16)f.y; dst[2] = (bf16)f.z; dst[3] = (bf16)f.w;
    }
    #pragma unroll
    for (int it = 0; it < 4; ++it) {
      int cc = tid + it * 256;             // 1024 float4 chunks for B
      int kk = cc >> 5;
      int nn = (cc & 31) << 2;
      const float4 f = *reinterpret_cast<const float4*>(
          Wd + (size_t)(kbase + kk) * N + n0 + nn);
      Bs[buf][nn + 0][kk] = (bf16)f.x;
      Bs[buf][nn + 1][kk] = (bf16)f.y;
      Bs[buf][nn + 2][kk] = (bf16)f.z;
      Bs[buf][nn + 3][kk] = (bf16)f.w;
    }
  };

  stage(0, 0);
  __syncthreads();

  for (int k0 = 0; k0 < K; k0 += 32) {
    const int cur = (k0 >> 5) & 1;
    const int nxt = cur ^ 1;

    // prefetch the k+64 tiles toward the WGP (global_prefetch_b8)
    if (k0 + 64 < K) {
      __builtin_prefetch(A + (size_t)(m0 + (tid >> 1)) * K + k0 + 64 + ((tid & 1) << 4), 0, 3);
      __builtin_prefetch(Wd + (size_t)(k0 + 64 + (tid >> 3)) * N + n0 + ((tid & 7) << 4), 0, 3);
    }
    // stage next tile into the alternate buffer while computing on `cur`
    if (k0 + 32 < K) stage(nxt, k0 + 32);

    // ---- build fragments per ISA 16-bit A(16x32)/B(32x16) layouts ----
    Frag fa[2], fb[4];
    #pragma unroll
    for (int i = 0; i < 2; ++i) {
      int ml = wm * 32 + i * 16 + lr;
      #pragma unroll
      for (int vv = 0; vv < 8; ++vv) {
        int kk = ((vv >> 2) << 4) + (hi << 3) + ((vv & 3) << 1);
        fa[i].u[vv] = *reinterpret_cast<const unsigned*>(&As[cur][ml][kk]);
      }
    }
    #pragma unroll
    for (int j = 0; j < 4; ++j) {
      int nl = wn * 64 + j * 16 + lr;
      int kb = hi << 4;
      #pragma unroll
      for (int vv = 0; vv < 8; ++vv)
        fb[j].u[vv] = *reinterpret_cast<const unsigned*>(&Bs[cur][nl][((vv >> 2) << 4) + (kb >> 1) + ((vv & 3) << 1)]);
    }

    // ---- 8 WMMAs per k-step ----
    #pragma unroll
    for (int i = 0; i < 2; ++i)
      #pragma unroll
      for (int j = 0; j < 4; ++j)
        acc[i][j] = __builtin_amdgcn_wmma_f32_16x16x32_bf16(
            false, fa[i].v, false, fb[j].v, (short)0, acc[i][j], false, false);

    // one barrier per k-step: protects both (a) next-iter compute reading
    // the freshly staged buffer and (b) staging that will overwrite `cur`
    // on the next iteration.
    __syncthreads();
  }

  // ---- epilogue: C layout (lane%16 = n, (lane/16)*8+v = m) ----
  #pragma unroll
  for (int i = 0; i < 2; ++i) {
    #pragma unroll
    for (int j = 0; j < 4; ++j) {
      #pragma unroll
      for (int vv = 0; vv < 8; ++vv) {
        int m = m0 + wm * 32 + i * 16 + (hi << 3) + vv;
        int n = n0 + wn * 64 + j * 16 + lr;
        float val = acc[i][j][vv];
        if (bias) val += bias[n];
        val *= scale;
        if (act == 1) val = gelu_f(val);
        if (m < M && n < N) C[(size_t)m * N + n] = val;
      }
    }
  }
}

// =====================================================================
// Sliding-window (band) attention + global-key columns.
// grid = (NC*16, H, B); each block: one chunk, 16 query rows, one head.
// =====================================================================
__global__ __launch_bounds__(256)
void k_band(const float* __restrict__ q, const float* __restrict__ kbuf,
            const float* __restrict__ vbuf, const int* __restrict__ sent,
            const int* __restrict__ s1, const int* __restrict__ e1,
            const int* __restrict__ s2, const int* __restrict__ e2,
            float* __restrict__ out)
{
  constexpr int R  = 16;           // query rows per block
  constexpr int J  = 3 * WIN + Gg; // 772 score columns
  constexpr int JP = 776;          // padded

  __shared__ __align__(16) float qs[R * DHh];  // 4 KB
  __shared__ float sc[R * JP];                 // ~49.7 KB
  __shared__ float red[256];

  const int tid = threadIdx.x;
  const int c   = blockIdx.x >> 4;          // chunk
  const int rq  = (blockIdx.x & 15) * R;    // row slice inside chunk
  const int h   = blockIdx.y;
  const int b   = blockIdx.z;

  const int gpos[4] = { s1[b], e1[b], s2[b], e2[b] };
  const size_t baseQ = ((size_t)b * Ssz + (size_t)c * WIN + rq) * Dm + (size_t)h * DHh;

  // load 16 query rows (already scaled by 1/sqrt(dh) in the q GEMM)
  for (int i = tid; i < R * DHh; i += 256) {
    int r = i >> 6, d = i & 63;
    qs[r * DHh + d] = q[baseQ + (size_t)r * Dm + d];
  }
  __syncthreads();

  // ---- band scores: thread owns score column j for all 16 rows ----
  for (int j = tid; j < 3 * WIN; j += 256) {
    int  kpos   = c * WIN - WIN + j;
    bool kvalid = (kpos >= 0 && kpos < Ssz);
    if (kvalid) {
      kvalid = (sent[b * Ssz + kpos] != 1);
      if (kvalid) {
        #pragma unroll
        for (int gg = 0; gg < Gg; ++gg) if (kpos == gpos[gg]) kvalid = false;
      }
    }
    float accv[R];
    #pragma unroll
    for (int r = 0; r < R; ++r) accv[r] = 0.f;
    if (kvalid) {
      const float* kr = kbuf + ((size_t)b * Ssz + kpos) * Dm + (size_t)h * DHh;
      for (int d0 = 0; d0 < DHh; d0 += 4) {
        const float4 kv4 = *reinterpret_cast<const float4*>(kr + d0);
        #pragma unroll
        for (int r = 0; r < R; ++r) {
          const float4 qv4 = *reinterpret_cast<const float4*>(&qs[r * DHh + d0]);
          accv[r] += qv4.x * kv4.x + qv4.y * kv4.y + qv4.z * kv4.z + qv4.w * kv4.w;
        }
      }
    }
    #pragma unroll
    for (int r = 0; r < R; ++r) {
      int  qi  = rq + r;
      int  rel = j - WIN - qi;
      bool ok  = kvalid && (rel >= -WIN) && (rel <= WIN);
      sc[r * JP + j] = ok ? accv[r] : -1e9f;
    }
  }
  // ---- global-key score columns (unmasked in reference) ----
  if (tid < Gg) {
    const int kpos = gpos[tid];
    const float* kr = kbuf + ((size_t)b * Ssz + kpos) * Dm + (size_t)h * DHh;
    for (int r = 0; r < R; ++r) {
      float dsum = 0.f;
      for (int d = 0; d < DHh; ++d) dsum += qs[r * DHh + d] * kr[d];
      sc[r * JP + 3 * WIN + tid] = dsum;
    }
  }
  __syncthreads();

  // ---- softmax: 16 threads per row ----
  const int r = tid >> 4, u = tid & 15;
  float mx = -1e30f;
  for (int j = u; j < J; j += 16) mx = fmaxf(mx, sc[r * JP + j]);
  red[tid] = mx; __syncthreads();
  if (u == 0) { float m = red[tid]; for (int t = 1; t < 16; ++t) m = fmaxf(m, red[tid + t]); red[tid] = m; }
  __syncthreads();
  const float rmax = red[r << 4];
  float se = 0.f;
  for (int j = u; j < J; j += 16) { float ev = __expf(sc[r * JP + j] - rmax); sc[r * JP + j] = ev; se += ev; }
  __syncthreads();
  red[tid] = se; __syncthreads();
  if (u == 0) { float s = 0.f; for (int t = 0; t < 16; ++t) s += red[tid + t]; red[tid] = s; }
  __syncthreads();
  const float inv = 1.0f / red[r << 4];
  for (int j = u; j < J; j += 16) sc[r * JP + j] *= inv;
  __syncthreads();

  // ---- out = probs @ V  (thread owns (row, dim) pairs) ----
  const int dd = tid & 63;
  const int rb = tid >> 6;
  for (int r2 = rb; r2 < R; r2 += 4) {
    const float* srow = &sc[r2 * JP];
    float accu = 0.f;
    for (int j = 0; j < 3 * WIN; ++j) {
      int kpos = c * WIN - WIN + j;
      if ((unsigned)kpos < (unsigned)Ssz)
        accu += srow[j] * vbuf[((size_t)b * Ssz + kpos) * Dm + (size_t)h * DHh + dd];
    }
    #pragma unroll
    for (int gg = 0; gg < Gg; ++gg)
      accu += srow[3 * WIN + gg] * vbuf[((size_t)b * Ssz + gpos[gg]) * Dm + (size_t)h * DHh + dd];
    out[baseQ + (size_t)r2 * Dm + dd] = accu;
  }
}

// =====================================================================
// qg = (x[g_idx] @ Wqg + bqg) * scale.   grid = B*G rows.
// =====================================================================
__global__ __launch_bounds__(256)
void k_qg(const float* __restrict__ x, const float* __restrict__ Wd,
          const float* __restrict__ bias,
          const int* __restrict__ s1, const int* __restrict__ e1,
          const int* __restrict__ s2, const int* __restrict__ e2,
          float* __restrict__ qg)
{
  __shared__ float xs[Dm];
  const int bg = blockIdx.x;
  const int b  = bg / Gg;
  const int g  = bg % Gg;
  const int pos = (g == 0) ? s1[b] : (g == 1) ? e1[b] : (g == 2) ? s2[b] : e2[b];
  for (int d = threadIdx.x; d < Dm; d += 256)
    xs[d] = x[((size_t)b * Ssz + pos) * Dm + d];
  __syncthreads();
  for (int n = threadIdx.x; n < Dm; n += 256) {
    float accu = bias[n];
    for (int kk = 0; kk < Dm; ++kk) accu += xs[kk] * Wd[(size_t)kk * Dm + n];
    qg[((size_t)b * Gg + g) * Dm + n] = accu * 0.125f;   // 1/sqrt(64)
  }
}

// =====================================================================
// Global-token full attention.  grid = (H, B, 2); 2 global rows / block.
// =====================================================================
__global__ __launch_bounds__(256)
void k_global(const float* __restrict__ qg, const float* __restrict__ kgf,
              const float* __restrict__ vgf, const int* __restrict__ sent,
              const int* __restrict__ s1, const int* __restrict__ e1,
              const int* __restrict__ s2, const int* __restrict__ e2,
              float* __restrict__ out)
{
  __shared__ float sc[2 * Ssz];    // 32 KB
  __shared__ float qgs[2 * DHh];
  __shared__ float red[256];

  const int h  = blockIdx.x;
  const int b  = blockIdx.y;
  const int g0 = blockIdx.z * 2;
  const int tid = threadIdx.x;
  const int allp[4] = { s1[b], e1[b], s2[b], e2[b] };

  if (tid < 2 * DHh) {
    int g = tid >> 6, d = tid & 63;
    qgs[g * DHh + d] = qg[((size_t)b * Gg + g0 + g) * Dm + (size_t)h * DHh + d];
  }
  __syncthreads();

  for (int s = tid; s < Ssz; s += 256) {
    const bool valid = (sent[b * Ssz + s] != 1);
    const float* kr = kgf + ((size_t)b * Ssz + s) * Dm + (size_t)h * DHh;
    #pragma unroll
    for (int g = 0; g < 2; ++g) {
      float dsum = 0.f;
      for (int d = 0; d < DHh; d += 4) {
        const float4 kv4 = *reinterpret_cast<const float4*>(kr + d);
        dsum += qgs[g * DHh + d]     * kv4.x + qgs[g * DHh + d + 1] * kv4.y
              + qgs[g * DHh + d + 2] * kv4.z + qgs[g * DHh + d + 3] * kv4.w;
      }
      sc[g * Ssz + s] = valid ? dsum : -1e9f;
    }
  }
  __syncthreads();

  // softmax: 128 threads per global row
  const int g = tid >> 7, u = tid & 127;
  float mx = -1e30f;
  for (int s = u; s < Ssz; s += 128) mx = fmaxf(mx, sc[g * Ssz + s]);
  red[tid] = mx; __syncthreads();
  for (int o = 64; o > 0; o >>= 1) { if (u < o) red[tid] = fmaxf(red[tid], red[tid + o]); __syncthreads(); }
  const float rmax = red[g << 7];
  __syncthreads();
  float se = 0.f;
  for (int s = u; s < Ssz; s += 128) { float ev = __expf(sc[g * Ssz + s] - rmax); sc[g * Ssz + s] = ev; se += ev; }
  red[tid] = se; __syncthreads();
  for (int o = 64; o > 0; o >>= 1) { if (u < o) red[tid] += red[tid + o]; __syncthreads(); }
  const float inv = 1.0f / red[g << 7];
  __syncthreads();
  for (int s = u; s < Ssz; s += 128) sc[g * Ssz + s] *= inv;
  __syncthreads();

  // out rows (overwrite band output at global positions)
  if (tid < 128) {
    const int gg = tid >> 6, dd = tid & 63;
    float accu = 0.f;
    for (int s = 0; s < Ssz; ++s)
      accu += sc[gg * Ssz + s] * vgf[((size_t)b * Ssz + s) * Dm + (size_t)h * DHh + dd];
    const int pos = allp[g0 + gg];
    out[((size_t)b * Ssz + pos) * Dm + (size_t)h * DHh + dd] = accu;
  }
}

// =====================================================================
// Span representation:  c = [m1, m2, m1*m2]  (6*D per batch)
// =====================================================================
__global__ __launch_bounds__(256)
void k_rep(const float* __restrict__ x,
           const int* __restrict__ s1, const int* __restrict__ e1,
           const int* __restrict__ s2, const int* __restrict__ e2,
           float* __restrict__ cvec)
{
  const int b = blockIdx.x;
  const int p1 = s1[b], q1 = e1[b], p2 = s2[b], q2 = e2[b];
  for (int d = threadIdx.x; d < Dm; d += 256) {
    const float sv1 = x[((size_t)b * Ssz + p1) * Dm + d];
    const float ev1 = x[((size_t)b * Ssz + q1) * Dm + d];
    const float sv2 = x[((size_t)b * Ssz + p2) * Dm + d];
    const float ev2 = x[((size_t)b * Ssz + q2) * Dm + d];
    float* cb = cvec + (size_t)b * 6 * Dm;
    cb[0 * Dm + d] = sv1;
    cb[1 * Dm + d] = ev1;
    cb[2 * Dm + d] = sv2;
    cb[3 * Dm + d] = ev2;
    cb[4 * Dm + d] = sv1 * sv2;
    cb[5 * Dm + d] = ev1 * ev2;
  }
}

// =====================================================================
// Small dense layer (head MLP).  grid = (B, N/256); act 1 = relu.
// =====================================================================
__global__ __launch_bounds__(256)
void k_dense(const float* __restrict__ in, const float* __restrict__ Wd,
             const float* __restrict__ bias, float* __restrict__ outp,
             int N, int K, int act)
{
  const int brow = blockIdx.x;
  const int n = blockIdx.y * 256 + threadIdx.x;
  if (n >= N) return;
  float accu = bias[n];
  const float* ar = in + (size_t)brow * K;
  for (int kk = 0; kk < K; ++kk) accu += ar[kk] * Wd[(size_t)kk * N + n];
  if (act == 1) accu = fmaxf(accu, 0.f);
  outp[(size_t)brow * N + n] = accu;
}

// =====================================================================
// logits -> sigmoid.  grid = B.
// =====================================================================
__global__ __launch_bounds__(256)
void k_final(const float* __restrict__ h2, const float* __restrict__ Wout,
             const float* __restrict__ bout, float* __restrict__ outp)
{
  __shared__ float red[256];
  const int b = blockIdx.x;
  float p = 0.f;
  for (int kk = threadIdx.x; kk < 2 * Dm; kk += 256)
    p += h2[(size_t)b * 2 * Dm + kk] * Wout[kk];
  red[threadIdx.x] = p; __syncthreads();
  for (int o = 128; o > 0; o >>= 1) { if (threadIdx.x < o) red[threadIdx.x] += red[threadIdx.x + o]; __syncthreads(); }
  if (threadIdx.x == 0) {
    const float z = red[0] + bout[0];
    outp[b] = 1.0f / (1.0f + __expf(-z));
  }
}

// ---------------- input flattening order (setup_inputs dict order) ----
enum {
  IN_SENT = 0, IN_S1, IN_E1, IN_S2, IN_E2,
  IN_WEMB, IN_PEMB, IN_TEMB, IN_ELG, IN_ELB,
  IN_WQ, IN_WK, IN_WV, IN_WO, IN_WQG, IN_WKG, IN_WVG,
  IN_BQ, IN_BK, IN_BV, IN_BO, IN_BQG, IN_BKG, IN_BVG, IN_BF2, IN_LN1B, IN_LN2B,
  IN_LN1G, IN_LN2G, IN_WF1, IN_BF1, IN_WF2,
  IN_W1, IN_B1, IN_W2, IN_B2, IN_WOUT, IN_BOUT
};

extern "C" void kernel_launch(void* const* d_in, const int* in_sizes, int n_in,
                              void* d_out, int out_size, void* d_ws, size_t ws_size,
                              hipStream_t stream)
{
  (void)in_sizes; (void)n_in; (void)out_size; (void)ws_size;

  const int* sent = (const int*)d_in[IN_SENT];
  const int* s1   = (const int*)d_in[IN_S1];
  const int* e1   = (const int*)d_in[IN_E1];
  const int* s2   = (const int*)d_in[IN_S2];
  const int* e2   = (const int*)d_in[IN_E2];
  const float* wemb = (const float*)d_in[IN_WEMB];
  const float* pemb = (const float*)d_in[IN_PEMB];
  const float* temb = (const float*)d_in[IN_TEMB];   // row 0 == type_emb[0]
  const float* elg  = (const float*)d_in[IN_ELG];
  const float* elb  = (const float*)d_in[IN_ELB];

  const float* Wq  = (const float*)d_in[IN_WQ];
  const float* Wk  = (const float*)d_in[IN_WK];
  const float* Wv  = (const float*)d_in[IN_WV];
  const float* Wo  = (const float*)d_in[IN_WO];
  const float* Wqg = (const float*)d_in[IN_WQG];
  const float* Wkg = (const float*)d_in[IN_WKG];
  const float* Wvg = (const float*)d_in[IN_WVG];
  const float* bq  = (const float*)d_in[IN_BQ];
  const float* bk  = (const float*)d_in[IN_BK];
  const float* bv  = (const float*)d_in[IN_BV];
  const float* bo  = (const float*)d_in[IN_BO];
  const float* bqg = (const float*)d_in[IN_BQG];
  const float* bkg = (const float*)d_in[IN_BKG];
  const float* bvg = (const float*)d_in[IN_BVG];
  const float* bf2 = (const float*)d_in[IN_BF2];
  const float* ln1b = (const float*)d_in[IN_LN1B];
  const float* ln2b = (const float*)d_in[IN_LN2B];
  const float* ln1g = (const float*)d_in[IN_LN1G];
  const float* ln2g = (const float*)d_in[IN_LN2G];
  const float* Wf1 = (const float*)d_in[IN_WF1];
  const float* bf1 = (const float*)d_in[IN_BF1];
  const float* Wf2 = (const float*)d_in[IN_WF2];

  const float* W1   = (const float*)d_in[IN_W1];
  const float* b1   = (const float*)d_in[IN_B1];
  const float* W2   = (const float*)d_in[IN_W2];
  const float* b2   = (const float*)d_in[IN_B2];
  const float* Wout = (const float*)d_in[IN_WOUT];
  const float* bout = (const float*)d_in[IN_BOUT];

  // ---------------- workspace layout (fp32) ----------------
  float* ws = (float*)d_ws;
  const size_t U = (size_t)BS * Dm;     // 6.29M floats per activation buffer
  float* x   = ws;
  float* qb  = x  + U;
  float* kb  = qb + U;
  float* vb  = kb + U;
  float* ao  = vb + U;                  // attention output (band + global)
  float* t0  = ao + U;                  // attn proj / ffn2 out
  float* ffh = t0 + U;                  // BS x FF
  float* qg  = ffh + (size_t)BS * FFd;
  float* cv  = qg + (size_t)Bb * Gg * Dm;
  float* h1  = cv + (size_t)Bb * 6 * Dm;
  float* h2  = h1 + (size_t)Bb * 2 * Dm;

  const dim3 gD(Dm / 128, BS / 128);          // (6, 64)
  const dim3 gF(FFd / 128, BS / 128);         // (24, 64)

  k_embed_ln<<<BS, 256, 0, stream>>>(sent, wemb, pemb, temb, elg, elb, x);

  for (int l = 0; l < NLn; ++l) {
    const size_t oDD = (size_t)l * Dm * Dm;
    const size_t oD  = (size_t)l * Dm;
    const float* lWq  = Wq  + oDD; const float* lbq  = bq  + oD;
    const float* lWk  = Wk  + oDD; const float* lbk  = bk  + oD;
    const float* lWv  = Wv  + oDD; const float* lbv  = bv  + oD;
    const float* lWo  = Wo  + oDD; const float* lbo  = bo  + oD;
    const float* lWqg = Wqg + oDD; const float* lbqg = bqg + oD;
    const float* lWkg = Wkg + oDD; const float* lbkg = bkg + oD;
    const float* lWvg = Wvg + oDD; const float* lbvg = bvg + oD;
    const float* lWf1 = Wf1 + (size_t)l * Dm * FFd;
    const float* lbf1 = bf1 + (size_t)l * FFd;
    const float* lWf2 = Wf2 + (size_t)l * FFd * Dm;
    const float* lbf2 = bf2 + oD;
    const float* l1g = ln1g + oD; const float* l1b = ln1b + oD;
    const float* l2g = ln2g + oD; const float* l2b = ln2b + oD;

    // q/k/v projections (q pre-scaled by 1/sqrt(dh))
    k_gemm<<<gD, 256, 0, stream>>>(x, lWq, lbq, qb, BS, Dm, Dm, 0.125f, 0);
    k_gemm<<<gD, 256, 0, stream>>>(x, lWk, lbk, kb, BS, Dm, Dm, 1.0f, 0);
    k_gemm<<<gD, 256, 0, stream>>>(x, lWv, lbv, vb, BS, Dm, Dm, 1.0f, 0);

    // sliding-window attention + global-key columns
    k_band<<<dim3(NC * (WIN / 16), Hh, Bb), 256, 0, stream>>>(
        qb, kb, vb, sent, s1, e1, s2, e2, ao);

    // global-token attention (reuses k/v buffers for kgf/vgf)
    k_gemm<<<gD, 256, 0, stream>>>(x, lWkg, lbkg, kb, BS, Dm, Dm, 1.0f, 0);
    k_gemm<<<gD, 256, 0, stream>>>(x, lWvg, lbvg, vb, BS, Dm, Dm, 1.0f, 0);
    k_qg<<<Bb * Gg, 256, 0, stream>>>(x, lWqg, lbqg, s1, e1, s2, e2, qg);
    k_global<<<dim3(Hh, Bb, Gg / 2), 256, 0, stream>>>(
        qg, kb, vb, sent, s1, e1, s2, e2, ao);

    // output projection + residual LN
    k_gemm<<<gD, 256, 0, stream>>>(ao, lWo, lbo, t0, BS, Dm, Dm, 1.0f, 0);
    k_ln_resid<<<BS, 256, 0, stream>>>(x, t0, l1g, l1b);

    // FFN (gelu fused into first GEMM epilogue) + residual LN
    k_gemm<<<gF, 256, 0, stream>>>(x, lWf1, lbf1, ffh, BS, FFd, Dm, 1.0f, 1);
    k_gemm<<<gD, 256, 0, stream>>>(ffh, lWf2, lbf2, t0, BS, Dm, FFd, 1.0f, 0);
    k_ln_resid<<<BS, 256, 0, stream>>>(x, t0, l2g, l2b);
  }

  // classification head
  k_rep<<<Bb, 256, 0, stream>>>(x, s1, e1, s2, e2, cv);
  k_dense<<<dim3(Bb, (2 * Dm) / 256), 256, 0, stream>>>(cv, W1, b1, h1, 2 * Dm, 6 * Dm, 1);
  k_dense<<<dim3(Bb, (2 * Dm) / 256), 256, 0, stream>>>(h1, W2, b2, h2, 2 * Dm, 2 * Dm, 1);
  k_final<<<Bb, 256, 0, stream>>>(h2, Wout, bout, (float*)d_out);
}
